// GeneratorBlock_36618891166275
// MI455X (gfx1250) — matrix operand order repaired and checked by reading, not compiled, not run on hardware
//
#include <hip/hip_runtime.h>
#include <hip/hip_bf16.h>
#include <math.h>

typedef unsigned short u16;
typedef unsigned int   u32;
typedef __attribute__((ext_vector_type(16))) __bf16 v16bf;
typedef __attribute__((ext_vector_type(8)))  float  v8f;

union FragB { v16bf v; u32 u[8]; };

__device__ __forceinline__ u16 f2bf(float x) {
  u32 u = __float_as_uint(x);
  u32 r = (u >> 16) & 1u;
  u += 0x7FFFu + r;                 // round-to-nearest-even
  return (u16)(u >> 16);
}

// K-index pattern for the 16-bit A fragment (16x32), per CDNA5 ISA 7.12.2:
// VGPR v<4  -> K = 2v + half*8 ; VGPR v>=4 -> K = 16+2(v-4) + half*8
__device__ __forceinline__ int kpat(int v, int half) {
  return (v < 4 ? 2 * v : 16 + 2 * (v - 4)) + half * 8;
}

// --- CDNA5 async memory->LDS DMA (tracked by ASYNCcnt), GV addressing -------
__device__ __forceinline__ void async_ldsb128(const void* gptr, void* lptr) {
  // low 32 bits of a generic shared pointer == workgroup-relative LDS offset
  u32 lds = (u32)(uintptr_t)lptr;
  asm volatile("global_load_async_to_lds_b128 %0, %1, off"
               :: "v"(lds), "v"(gptr) : "memory");
}
__device__ __forceinline__ void wait_async0() {
#if __has_builtin(__builtin_amdgcn_s_wait_asynccnt)
  __builtin_amdgcn_s_wait_asynccnt(0);
#else
  asm volatile("s_wait_asynccnt 0x0" ::: "memory");
#endif
}

// ---------------------------------------------------------------------------
// f32 -> bf16 elementwise convert (weights / plan)
// ---------------------------------------------------------------------------
__global__ void f2b_kernel(const float* __restrict__ in, u16* __restrict__ out, int n) {
  int i = (blockIdx.x * blockDim.x + threadIdx.x) * 4;
  if (i + 3 < n) {
    float4 v = *(const float4*)(in + i);
    out[i + 0] = f2bf(v.x);
    out[i + 1] = f2bf(v.y);
    out[i + 2] = f2bf(v.z);
    out[i + 3] = f2bf(v.w);
  }
}

// ---------------------------------------------------------------------------
// LayerNorm over C=1024, fp32 in -> bf16 out. One block (256 thr) per row.
// ---------------------------------------------------------------------------
__global__ __launch_bounds__(256) void ln_bf16_kernel(
    const float* __restrict__ x, const float* __restrict__ g,
    const float* __restrict__ bb, u16* __restrict__ out) {
  const int row  = blockIdx.x;
  const int tid  = threadIdx.x;
  const int lane = tid & 31;
  const int wave = tid >> 5;
  const float* xr = x + (size_t)row * 1024;
  float4 v = ((const float4*)xr)[tid];
  float sum = v.x + v.y + v.z + v.w;
  float sq  = v.x * v.x + v.y * v.y + v.z * v.z + v.w * v.w;
#pragma unroll
  for (int o = 16; o > 0; o >>= 1) {
    sum += __shfl_xor(sum, o, 32);
    sq  += __shfl_xor(sq,  o, 32);
  }
  __shared__ float s1[8], s2[8];
  __shared__ float smean, srstd;
  if (lane == 0) { s1[wave] = sum; s2[wave] = sq; }
  __syncthreads();
  if (tid == 0) {
    float a = 0.f, c2 = 0.f;
#pragma unroll
    for (int i = 0; i < 8; ++i) { a += s1[i]; c2 += s2[i]; }
    float mean = a * (1.0f / 1024.0f);
    float var  = c2 * (1.0f / 1024.0f) - mean * mean;
    smean = mean;
    srstd = rsqrtf(var + 1e-5f);
  }
  __syncthreads();
  float mean = smean, rstd = srstd;
  float4 gv = ((const float4*)g)[tid];
  float4 bv = ((const float4*)bb)[tid];
  u16* orow = out + (size_t)row * 1024;
  int i0 = tid * 4;
  orow[i0 + 0] = f2bf((v.x - mean) * rstd * gv.x + bv.x);
  orow[i0 + 1] = f2bf((v.y - mean) * rstd * gv.y + bv.y);
  orow[i0 + 2] = f2bf((v.z - mean) * rstd * gv.z + bv.z);
  orow[i0 + 3] = f2bf((v.w - mean) * rstd * gv.w + bv.w);
}

// ---------------------------------------------------------------------------
// Tiled bf16 WMMA GEMM: out[M,N] = A[M,K] @ Bw[K,N] + bias (+ epilogue)
// Block tile 64x128, 8 waves, each wave 32x32 (2x2 v_wmma tiles).
// K stepped by 64, async (ASYNCcnt) double-buffered LDS staging:
// tile t+1 DMAs memory->LDS while tile t's 8 WMMAs issue; one barrier/step.
// MODE 0: bf16 out;  MODE 1: exact-GELU -> bf16 out;  MODE 2: fp32 out = resid + y
// ---------------------------------------------------------------------------
template <int MODE>
__global__ __launch_bounds__(256) void gemm_bf16_kernel(
    const u16* __restrict__ A, const u16* __restrict__ Bw,
    const float* __restrict__ bias, const float* __restrict__ resid,
    float* __restrict__ outf, u16* __restrict__ outb, int M, int N, int K) {
  constexpr int ASTR = 72;   // 144 B row stride (16B aligned, conflict-free frags)
  constexpr int BSTR = 136;  // 272 B row stride
  __shared__ u16 As[2][64 * ASTR];   // 64 rows x 64 K  (x2 buffers)
  __shared__ u16 Bs[2][64 * BSTR];   // 64 K   x 128 N  (x2 buffers)

  const int tid  = threadIdx.x;
  const int lane = tid & 31;
  const int wave = tid >> 5;
  const int wm   = wave >> 2;   // 0..1
  const int wn   = wave & 3;    // 0..3
  const int m0   = blockIdx.y * 64;
  const int n0   = blockIdx.x * 128;
  const int half = lane >> 4;
  const int r    = lane & 15;

  const v8f zf = {0.f, 0.f, 0.f, 0.f, 0.f, 0.f, 0.f, 0.f};
  v8f acc[2][2];
  acc[0][0] = zf; acc[0][1] = zf; acc[1][0] = zf; acc[1][1] = zf;

  const int ar = tid >> 2, ac = (tid & 3) * 16;  // A: 64x64, 32B per thread
  const int br = tid >> 2, bc = (tid & 3) * 32;  // B: 64x128, 64B per thread

  const u16* agp = A  + (size_t)(m0 + ar) * K + ac;
  const u16* bgp = Bw + (size_t)br * N + n0 + bc;

  auto stage = [&](int buf, int k0) {
    const u16* ag = agp + k0;
    async_ldsb128(ag,     &As[buf][ar * ASTR + ac]);
    async_ldsb128(ag + 8, &As[buf][ar * ASTR + ac + 8]);
    const u16* bg = bgp + (size_t)k0 * N;
    async_ldsb128(bg,      &Bs[buf][br * BSTR + bc]);
    async_ldsb128(bg + 8,  &Bs[buf][br * BSTR + bc + 8]);
    async_ldsb128(bg + 16, &Bs[buf][br * BSTR + bc + 16]);
    async_ldsb128(bg + 24, &Bs[buf][br * BSTR + bc + 24]);
  };

  const int nk = K >> 6;
  stage(0, 0);
  wait_async0();
  __syncthreads();

  for (int kt = 0; kt < nk; ++kt) {
    const int cur = kt & 1;
    if (kt + 1 < nk) stage((kt + 1) & 1, (kt + 1) << 6);  // prefetch DMA

#pragma unroll
    for (int ks = 0; ks < 2; ++ks) {  // two 32-deep sub-steps of this tile
      FragB af[2], bf[2];
#pragma unroll
      for (int tm = 0; tm < 2; ++tm) {
        const u16* abase = &As[cur][(wm * 32 + tm * 16 + r) * ASTR + ks * 32];
#pragma unroll
        for (int v = 0; v < 8; ++v) af[tm].u[v] = *(const u32*)(abase + kpat(v, half));
      }
#pragma unroll
      for (int tn = 0; tn < 2; ++tn) {
        const u16* bbase = &Bs[cur][(ks * 32) * BSTR + wn * 32 + tn * 16 + r];
#pragma unroll
        for (int v = 0; v < 8; ++v) bf[tn].u[v] = (u32)bbase[kpat(v, half) * BSTR] | ((u32)bbase[(kpat(v, half) + 1) * BSTR] << 16);
      }
#pragma unroll
      for (int tm = 0; tm < 2; ++tm)
#pragma unroll
        for (int tn = 0; tn < 2; ++tn)
          acc[tm][tn] = __builtin_amdgcn_wmma_f32_16x16x32_bf16(
              false, af[tm].v, false, bf[tn].v, (short)0, acc[tm][tn], false, false);
    }
    wait_async0();
    __syncthreads();
  }

#pragma unroll
  for (int tm = 0; tm < 2; ++tm) {
#pragma unroll
    for (int tn = 0; tn < 2; ++tn) {
#pragma unroll
      for (int v = 0; v < 8; ++v) {
        int row = m0 + wm * 32 + tm * 16 + v + 8 * half;
        int col = n0 + wn * 32 + tn * 16 + r;
        float val = acc[tm][tn][v] + bias[col];
        if (MODE == 1) val = 0.5f * val * (1.0f + erff(val * 0.70710678118654752f));
        size_t idx = (size_t)row * N + col;
        if (MODE <= 1) outb[idx] = f2bf(val);
        else           outf[idx] = resid[idx] + val;
      }
    }
  }
}

// ---------------------------------------------------------------------------
// Flash-attention with WMMA. 128 threads = 4 waves; each wave owns 16 q rows,
// streaming 64-key tiles. D = 64. Online softmax, causal optional.
// ---------------------------------------------------------------------------
__global__ __launch_bounds__(128) void attn_wmma_kernel(
    const u16* __restrict__ qp, const u16* __restrict__ kp, const u16* __restrict__ vp,
    u16* __restrict__ op, int q_stride, int kv_stride, int out_stride,
    int q_off, int k_off, int v_off, int Tq, int Tk, int causal, float scale) {
  constexpr int KTS = 72, VS = 72, PSTR = 72;
  __shared__ u16 Kt[64 * KTS];        // K tile transposed: [d][key]
  __shared__ u16 Vsm[64 * VS];        // V tile: [key][d]
  __shared__ u16 Ps[4][16 * PSTR];    // per-wave P tile: [qrow][key]

  const int tid  = threadIdx.x;
  const int lane = tid & 31;
  const int wave = tid >> 5;
  const int h    = blockIdx.y;
  const int b    = blockIdx.z;
  const int qblk = blockIdx.x * 64;
  const int q0   = qblk + wave * 16;
  const int half = lane >> 4;
  const int r    = lane & 15;

  const u16* qbase = qp + (size_t)(b * Tq) * q_stride + q_off + h * 64;
  const u16* kbase = kp + (size_t)(b * Tk) * kv_stride + k_off + h * 64;
  const u16* vbase = vp + (size_t)(b * Tk) * kv_stride + v_off + h * 64;

  // Q fragments (16 rows x 64 d = two 16x32 A frags), loaded straight from global
  FragB qf[2];
#pragma unroll
  for (int f = 0; f < 2; ++f)
#pragma unroll
    for (int v = 0; v < 8; ++v)
      qf[f].u[v] = *(const u32*)(qbase + (size_t)(q0 + r) * q_stride + f * 32 + kpat(v, half));

  const v8f zf = {0.f, 0.f, 0.f, 0.f, 0.f, 0.f, 0.f, 0.f};
  v8f O[4];
#pragma unroll
  for (int dt = 0; dt < 4; ++dt) O[dt] = zf;
  float mi[8], li[8];
#pragma unroll
  for (int v = 0; v < 8; ++v) { mi[v] = -1e30f; li[v] = 0.f; }

  int kend = causal ? (qblk + 64 < Tk ? qblk + 64 : Tk) : Tk;

  for (int kb = 0; kb < kend; kb += 64) {
    {  // cooperative stage: K transposed + V straight (64 keys x 64 d)
      int key = tid >> 1;
      int dh  = (tid & 1) * 32;
      const u16* kg = kbase + (size_t)(kb + key) * kv_stride + dh;
      u16 tmp[32];
      *(uint4*)&tmp[0]  = *(const uint4*)(kg);
      *(uint4*)&tmp[8]  = *(const uint4*)(kg + 8);
      *(uint4*)&tmp[16] = *(const uint4*)(kg + 16);
      *(uint4*)&tmp[24] = *(const uint4*)(kg + 24);
#pragma unroll
      for (int j = 0; j < 32; ++j) Kt[(dh + j) * KTS + key] = tmp[j];
      const u16* vg = vbase + (size_t)(kb + key) * kv_stride + dh;
      uint4* vd = (uint4*)&Vsm[key * VS + dh];
      vd[0] = *(const uint4*)(vg);
      vd[1] = *(const uint4*)(vg + 8);
      vd[2] = *(const uint4*)(vg + 16);
      vd[3] = *(const uint4*)(vg + 24);
    }
    __syncthreads();

    // S = Q @ K^T : 16 x 64 (four 16x16 tiles)
    v8f S[4];
#pragma unroll
    for (int nt = 0; nt < 4; ++nt) {
      v8f acc = zf;
#pragma unroll
      for (int f = 0; f < 2; ++f) {
        FragB bfr;
        const u16* bbase = &Kt[(f * 32) * KTS + nt * 16 + r];
#pragma unroll
        for (int v = 0; v < 8; ++v) bfr.u[v] = (u32)bbase[kpat(v, half) * KTS] | ((u32)bbase[(kpat(v, half) + 1) * KTS] << 16);
        acc = __builtin_amdgcn_wmma_f32_16x16x32_bf16(
            false, qf[f].v, false, bfr.v, (short)0, acc, false, false);
      }
      S[nt] = acc;
    }

    // online softmax per row (row = q0 + v + 8*half; row spread over 16 lanes)
#pragma unroll
    for (int v = 0; v < 8; ++v) {
      int qrow = q0 + v + 8 * half;
      float rmax = -1e30f;
#pragma unroll
      for (int nt = 0; nt < 4; ++nt) {
        float s = S[nt][v] * scale;
        int kcol = kb + nt * 16 + r;
        if (causal && kcol > qrow) s = -1e30f;
        S[nt][v] = s;
        rmax = fmaxf(rmax, s);
      }
#pragma unroll
      for (int o = 8; o > 0; o >>= 1) rmax = fmaxf(rmax, __shfl_xor(rmax, o, 32));
      float mn = fmaxf(mi[v], rmax);
      float sc = __expf(mi[v] - mn);
      float rsum = 0.f;
#pragma unroll
      for (int nt = 0; nt < 4; ++nt) {
        float p = __expf(S[nt][v] - mn);
        S[nt][v] = p;
        rsum += p;
      }
#pragma unroll
      for (int o = 8; o > 0; o >>= 1) rsum += __shfl_xor(rsum, o, 32);
      li[v] = li[v] * sc + rsum;
      mi[v] = mn;
#pragma unroll
      for (int dt = 0; dt < 4; ++dt) O[dt][v] *= sc;
#pragma unroll
      for (int nt = 0; nt < 4; ++nt)
        Ps[wave][(v + 8 * half) * PSTR + nt * 16 + r] = f2bf(S[nt][v]);
    }
    __syncthreads();

    // O += P @ V   (P via LDS round-trip to get A-fragment layout)
#pragma unroll
    for (int dt = 0; dt < 4; ++dt) {
#pragma unroll
      for (int f = 0; f < 2; ++f) {
        FragB pa, vb;
        const u16* pbase = &Ps[wave][r * PSTR + f * 32];
#pragma unroll
        for (int v = 0; v < 8; ++v) pa.u[v] = *(const u32*)(pbase + kpat(v, half));
        const u16* vb2 = &Vsm[(f * 32) * VS + dt * 16 + r];
#pragma unroll
        for (int v = 0; v < 8; ++v) vb.u[v] = (u32)vb2[kpat(v, half) * VS] | ((u32)vb2[(kpat(v, half) + 1) * VS] << 16);
        O[dt] = __builtin_amdgcn_wmma_f32_16x16x32_bf16(
            false, pa.v, false, vb.v, (short)0, O[dt], false, false);
      }
    }
    __syncthreads();
  }

  u16* obase = op + (size_t)(b * Tq) * out_stride + h * 64;
#pragma unroll
  for (int v = 0; v < 8; ++v) {
    float inv = 1.0f / li[v];
    int row = q0 + v + 8 * half;
#pragma unroll
    for (int dt = 0; dt < 4; ++dt)
      obase[(size_t)row * out_stride + dt * 16 + r] = f2bf(O[dt][v] * inv);
  }
}

// ---------------------------------------------------------------------------
extern "C" void kernel_launch(void* const* d_in, const int* in_sizes, int n_in,
                              void* d_out, int out_size, void* d_ws, size_t ws_size,
                              hipStream_t stream) {
  (void)in_sizes; (void)n_in; (void)out_size; (void)ws_size;
  const int Bb = 2, T = 2048, TP = 512, C = 1024, H = 16;
  const int M  = Bb * T;    // 4096
  const int MP = Bb * TP;   // 1024

  const float* x           = (const float*)d_in[0];
  const float* plan        = (const float*)d_in[1];
  const float* ln1_g       = (const float*)d_in[2];
  const float* ln1_b       = (const float*)d_in[3];
  const float* c_attn_w    = (const float*)d_in[4];
  const float* c_attn_b    = (const float*)d_in[5];
  const float* attn_proj_w = (const float*)d_in[6];
  const float* attn_proj_b = (const float*)d_in[7];
  const float* ln2_g       = (const float*)d_in[8];
  const float* ln2_b       = (const float*)d_in[9];
  const float* q_w         = (const float*)d_in[10];
  const float* q_b         = (const float*)d_in[11];
  const float* k_w         = (const float*)d_in[12];
  const float* k_b         = (const float*)d_in[13];
  const float* v_w         = (const float*)d_in[14];
  const float* v_b         = (const float*)d_in[15];
  const float* ca_proj_w   = (const float*)d_in[16];
  const float* ca_proj_b   = (const float*)d_in[17];
  const float* ln3_g       = (const float*)d_in[18];
  const float* ln3_b       = (const float*)d_in[19];
  const float* fc_w        = (const float*)d_in[20];
  const float* fc_b        = (const float*)d_in[21];
  const float* mlp_proj_w  = (const float*)d_in[22];
  const float* mlp_proj_b  = (const float*)d_in[23];
  float* out = (float*)d_out;

  char* ws = (char*)d_ws;
  size_t off = 0;
  auto alloc = [&](size_t elems, size_t esz) -> void* {
    void* p = ws + off;
    off = (off + elems * esz + 255) & ~(size_t)255;
    return p;
  };

  u16* wb_cattn  = (u16*)alloc((size_t)C * 3 * C, 2);
  u16* wb_aproj  = (u16*)alloc((size_t)C * C, 2);
  u16* wb_q      = (u16*)alloc((size_t)C * C, 2);
  u16* wb_k      = (u16*)alloc((size_t)C * C, 2);
  u16* wb_v      = (u16*)alloc((size_t)C * C, 2);
  u16* wb_caproj = (u16*)alloc((size_t)C * C, 2);
  u16* wb_fc     = (u16*)alloc((size_t)C * 4 * C, 2);
  u16* wb_mproj  = (u16*)alloc((size_t)4 * C * C, 2);
  u16* plan_b    = (u16*)alloc((size_t)MP * C, 2);
  u16* ln_b      = (u16*)alloc((size_t)M * C, 2);
  u16* qkv_b     = (u16*)alloc((size_t)M * 3 * C, 2);
  u16* attno_b   = (u16*)alloc((size_t)M * C, 2);
  u16* qca_b     = (u16*)alloc((size_t)M * C, 2);
  u16* kpl_b     = (u16*)alloc((size_t)MP * C, 2);
  u16* vpl_b     = (u16*)alloc((size_t)MP * C, 2);
  u16* h_b       = (u16*)alloc((size_t)M * 4 * C, 2);
  float* x1      = (float*)alloc((size_t)M * C, 4);
  float* x2      = (float*)alloc((size_t)M * C, 4);

  auto conv = [&](const float* src, u16* dst, size_t n) {
    f2b_kernel<<<dim3((unsigned)(n / 1024)), 256, 0, stream>>>(src, dst, (int)n);
  };
  conv(c_attn_w,    wb_cattn,  (size_t)C * 3 * C);
  conv(attn_proj_w, wb_aproj,  (size_t)C * C);
  conv(q_w,         wb_q,      (size_t)C * C);
  conv(k_w,         wb_k,      (size_t)C * C);
  conv(v_w,         wb_v,      (size_t)C * C);
  conv(ca_proj_w,   wb_caproj, (size_t)C * C);
  conv(fc_w,        wb_fc,     (size_t)C * 4 * C);
  conv(mlp_proj_w,  wb_mproj,  (size_t)4 * C * C);
  conv(plan,        plan_b,    (size_t)MP * C);

  const float scale = 0.125f;  // 1/sqrt(64)

  // x1 = x + self_attn(ln1(x))
  ln_bf16_kernel<<<M, 256, 0, stream>>>(x, ln1_g, ln1_b, ln_b);
  gemm_bf16_kernel<0><<<dim3(3 * C / 128, M / 64), 256, 0, stream>>>(
      ln_b, wb_cattn, c_attn_b, nullptr, nullptr, qkv_b, M, 3 * C, C);
  attn_wmma_kernel<<<dim3(T / 64, H, Bb), 128, 0, stream>>>(
      qkv_b, qkv_b, qkv_b, attno_b, 3 * C, 3 * C, C, 0, C, 2 * C, T, T, 1, scale);
  gemm_bf16_kernel<2><<<dim3(C / 128, M / 64), 256, 0, stream>>>(
      attno_b, wb_aproj, attn_proj_b, x, x1, nullptr, M, C, C);

  // x2 = x1 + cross_attn(ln2(x1), plan)
  ln_bf16_kernel<<<M, 256, 0, stream>>>(x1, ln2_g, ln2_b, ln_b);
  gemm_bf16_kernel<0><<<dim3(C / 128, M / 64), 256, 0, stream>>>(
      ln_b, wb_q, q_b, nullptr, nullptr, qca_b, M, C, C);
  gemm_bf16_kernel<0><<<dim3(C / 128, MP / 64), 256, 0, stream>>>(
      plan_b, wb_k, k_b, nullptr, nullptr, kpl_b, MP, C, C);
  gemm_bf16_kernel<0><<<dim3(C / 128, MP / 64), 256, 0, stream>>>(
      plan_b, wb_v, v_b, nullptr, nullptr, vpl_b, MP, C, C);
  attn_wmma_kernel<<<dim3(T / 64, H, Bb), 128, 0, stream>>>(
      qca_b, kpl_b, vpl_b, attno_b, C, C, C, 0, 0, 0, T, TP, 0, scale);
  gemm_bf16_kernel<2><<<dim3(C / 128, M / 64), 256, 0, stream>>>(
      attno_b, wb_caproj, ca_proj_b, x1, x2, nullptr, M, C, C);

  // out = x2 + mlp(ln3(x2))
  ln_bf16_kernel<<<M, 256, 0, stream>>>(x2, ln3_g, ln3_b, ln_b);
  gemm_bf16_kernel<1><<<dim3(4 * C / 128, M / 64), 256, 0, stream>>>(
      ln_b, wb_fc, fc_b, nullptr, nullptr, h_b, M, 4 * C, C);
  gemm_bf16_kernel<2><<<dim3(C / 128, M / 64), 256, 0, stream>>>(
      h_b, wb_mproj, mlp_proj_b, x2, out, nullptr, M, C, 4 * C);
}
